// GPTQLinear_33887291965473
// MI455X (gfx1250) — matrix operand order repaired and checked by reading, not compiled, not run on hardware
//
#include <hip/hip_runtime.h>
#include <hip/hip_bf16.h>

#define OUT_F 11008
#define IN_F  4096
#define QW_STRIDE (IN_F / 2)   /* 2048 int32 per output row */
#define NG 32
#define M_ROWS 32

typedef __attribute__((ext_vector_type(16))) _Float16 v16h;
typedef __attribute__((ext_vector_type(8)))  _Float16 v8h;
typedef __attribute__((ext_vector_type(8)))  float    v8f;
typedef __attribute__((ext_vector_type(4)))  unsigned int u32x4;

union VFrag {
    v16h v;
    v8h  p[2];
    _Float16 h[16];
};

// Kernel 1: gather x by perm and convert fp32 -> f16 (xp[m][j] = x[m][perm[j]])
__global__ void prep_x_kernel(const float* __restrict__ x,
                              const int* __restrict__ perm,
                              _Float16* __restrict__ xp) {
    int j = blockIdx.x * blockDim.x + threadIdx.x;  // 0 .. 32*4096-1
    int m = j >> 12;
    int c = j & (IN_F - 1);
    xp[j] = (_Float16)x[m * IN_F + perm[c]];
}

// Main kernel: 344 blocks x 256 threads (8 waves).
// wave -> (ntile in {0,1}, ksplit in {0..3}); each wave: 16 N-cols, 1024 K.
template <bool PRE>
__global__ __launch_bounds__(256)
void gptq_wmma_kernel(const float* __restrict__ x,
                      const unsigned int* __restrict__ qw,
                      const float* __restrict__ scales,
                      const float* __restrict__ zps,
                      const float* __restrict__ bias,
                      const int* __restrict__ perm,
                      const _Float16* __restrict__ xp,
                      float* __restrict__ out) {
    __shared__ float red[2][3][512];  // [ntile][source wave][lane*16 + i]

    const int lane = threadIdx.x & 31;
    const int wave = threadIdx.x >> 5;
    const int ntl  = wave >> 2;       // 0..1: which 16-col N tile
    const int ks   = wave & 3;        // 0..3: K split
    const int lrow = lane & 15;
    const int hi   = lane >> 4;       // lane half selects K window
    const int n    = blockIdx.x * 32 + ntl * 16 + lrow;

    const int kwave = ks * (IN_F / 4);  // 1024 K per wave

    // B stream: two runs of 4 int32 (K 8*hi..+7 and 16+8*hi..+7) per lane per step
    const unsigned int* qptr =
        qw + (size_t)n * QW_STRIDE + (kwave >> 1) + hi * 4;
    // A stream (PRE path): f16 rows of permuted x
    const _Float16* aptr0 = PRE ? (xp + (size_t)lrow * IN_F + kwave + hi * 8) : nullptr;
    const _Float16* aptr1 = PRE ? (aptr0 + 16 * IN_F) : nullptr;

    v8f c0 = {};
    v8f c1 = {};

    for (int gi = 0; gi < 8; ++gi) {           // 8 groups of 128 per wave
        const int g  = ks * 8 + gi;
        const float s  = scales[(size_t)n * NG + g];
        const float zz = zps[(size_t)n * NG + g] - 8.0f * s;

        // stream-ahead hint for the weight stream (~1KB ahead)
        __builtin_prefetch((const void*)(qptr + 256), 0, 1);

#pragma unroll
        for (int u = 0; u < 4; ++u) {          // 4 WMMA K-steps per group
            // ---- B fragment: dequant 16 nibbles -> f16 ----
            u32x4 q0 = __builtin_nontemporal_load((const u32x4*)qptr);
            u32x4 q1 = __builtin_nontemporal_load((const u32x4*)(qptr + 8));
            qptr += 16;
            unsigned int qa[8] = {q0.x, q0.y, q0.z, q0.w, q1.x, q1.y, q1.z, q1.w};
            VFrag b;
#pragma unroll
            for (int p = 0; p < 8; ++p) {
                unsigned int q = qa[p];
                // payload is one byte per int32 (values 0..255), so the high
                // nibble is exactly (q >> 4): no mask needed, and
                // v_cvt_f32_ubyte0 only reads bits [7:0] anyway.
                float w0 = (float)(q & 15u) * s + zz;  // even K (low nibble)
                float w1 = (float)(q >> 4) * s + zz;   // odd K (high nibble)
                b.h[2 * p]     = (_Float16)w0;
                b.h[2 * p + 1] = (_Float16)w1;
            }

            // ---- A fragments (two 16-row M tiles share B) ----
            VFrag a0, a1;
            if (PRE) {
                a0.p[0] = *(const v8h*)aptr0;         // K base+hi*8 .. +7
                a0.p[1] = *(const v8h*)(aptr0 + 16);  // K base+hi*8+16 .. +23
                a1.p[0] = *(const v8h*)aptr1;
                a1.p[1] = *(const v8h*)(aptr1 + 16);
                aptr0 += 32;
                aptr1 += 32;
            } else {
                const int kbase = kwave + gi * 128 + u * 32;
#pragma unroll
                for (int c = 0; c < 8; ++c) {
                    int col0 = kbase + hi * 8 + c;
                    int pc0 = perm[col0];
                    int pc1 = perm[col0 + 16];
                    a0.h[c]     = (_Float16)x[(size_t)lrow * IN_F + pc0];
                    a0.h[8 + c] = (_Float16)x[(size_t)lrow * IN_F + pc1];
                    a1.h[c]     = (_Float16)x[(size_t)(lrow + 16) * IN_F + pc0];
                    a1.h[8 + c] = (_Float16)x[(size_t)(lrow + 16) * IN_F + pc1];
                }
            }

            // D = A x B + C  (f32 accumulate)
            c0 = __builtin_amdgcn_wmma_f32_16x16x32_f16(
                false, a0.v, false, b.v, (short)0, c0, false, false);
            c1 = __builtin_amdgcn_wmma_f32_16x16x32_f16(
                false, a1.v, false, b.v, (short)0, c1, false, false);
        }
    }

    // ---- cross-wave K reduction through LDS ----
    if (ks != 0) {
#pragma unroll
        for (int i = 0; i < 8; ++i) {
            red[ntl][ks - 1][lane * 16 + i]     = c0[i];
            red[ntl][ks - 1][lane * 16 + 8 + i] = c1[i];
        }
    }
    __syncthreads();
    if (ks == 0) {
#pragma unroll
        for (int w = 0; w < 3; ++w) {
#pragma unroll
            for (int i = 0; i < 8; ++i) {
                c0[i] += red[ntl][w][lane * 16 + i];
                c1[i] += red[ntl][w][lane * 16 + 8 + i];
            }
        }
        const float bv = bias[n];
        // C layout: VGPR i, lane half hi -> m = i + 8*hi (tile0), +16 (tile1)
#pragma unroll
        for (int i = 0; i < 8; ++i) {
            int m = i + 8 * hi;
            __builtin_nontemporal_store(c0[i] + bv, &out[(size_t)m * OUT_F + n]);
            __builtin_nontemporal_store(c1[i] + bv, &out[(size_t)(m + 16) * OUT_F + n]);
        }
    }
}

extern "C" void kernel_launch(void* const* d_in, const int* in_sizes, int n_in,
                              void* d_out, int out_size, void* d_ws, size_t ws_size,
                              hipStream_t stream) {
    const float* x      = (const float*)d_in[0];
    const unsigned int* qw = (const unsigned int*)d_in[1];
    const float* scales = (const float*)d_in[2];
    const float* zps    = (const float*)d_in[3];
    const float* bias   = (const float*)d_in[4];
    const int*   perm   = (const int*)d_in[5];
    float* out = (float*)d_out;

    const size_t xp_bytes = (size_t)M_ROWS * IN_F * sizeof(_Float16);  // 256 KB
    dim3 block(256);
    dim3 grid(OUT_F / 32);  // 344 blocks: 2 N-tiles x 4 K-splits each

    if (ws_size >= xp_bytes) {
        _Float16* xp = (_Float16*)d_ws;
        prep_x_kernel<<<dim3((M_ROWS * IN_F) / 256), block, 0, stream>>>(x, perm, xp);
        gptq_wmma_kernel<true><<<grid, block, 0, stream>>>(
            x, qw, scales, zps, bias, perm, xp, out);
    } else {
        gptq_wmma_kernel<false><<<grid, block, 0, stream>>>(
            x, qw, scales, zps, bias, perm, nullptr, out);
    }
}
